// Attention_4063039062503
// MI455X (gfx1250) — matrix operand order repaired and checked by reading, not compiled, not run on hardware
//
#include <hip/hip_runtime.h>
#include <hip/hip_bf16.h>

typedef __attribute__((ext_vector_type(16))) _Float16 v16h;
typedef __attribute__((ext_vector_type(8)))  float    v8f;
typedef __attribute__((ext_vector_type(4)))  int      v4i;

#define C_DIM 128
#define N_DIM 4096
#define B_DIM 4
#define NT    64      // n-tile per attention iteration

struct U32x8 { uint4 a, b; };
struct H16   { _Float16 h[16]; };
struct H8    { _Float16 h[8]; };
struct H4    { _Float16 h[4]; };

__device__ __forceinline__ v16h make_v16h(uint4 lo, uint4 hi) {
    U32x8 u{lo, hi};
    return __builtin_bit_cast(v16h, u);
}

__device__ __forceinline__ v16h cvt16(float4 a, float4 b, float4 c, float4 d) {
    H16 r;
    r.h[0]=(_Float16)a.x; r.h[1]=(_Float16)a.y; r.h[2]=(_Float16)a.z; r.h[3]=(_Float16)a.w;
    r.h[4]=(_Float16)b.x; r.h[5]=(_Float16)b.y; r.h[6]=(_Float16)b.z; r.h[7]=(_Float16)b.w;
    r.h[8]=(_Float16)c.x; r.h[9]=(_Float16)c.y; r.h[10]=(_Float16)c.z; r.h[11]=(_Float16)c.w;
    r.h[12]=(_Float16)d.x; r.h[13]=(_Float16)d.y; r.h[14]=(_Float16)d.z; r.h[15]=(_Float16)d.w;
    return __builtin_bit_cast(v16h, r);
}

__device__ __forceinline__ v8f wmma_f16(v16h a, v16h b, v8f c) {
    return __builtin_amdgcn_wmma_f32_16x16x32_f16(false, a, false, b, (short)0, c, false, false);
}

// ---- CDNA5 async global->LDS copy (16B per lane), ASYNCcnt-tracked ----
__device__ __forceinline__ void async_copy16(_Float16* lds, const _Float16* gsrc) {
#if __has_builtin(__builtin_amdgcn_global_load_async_to_lds_b128)
    __builtin_amdgcn_global_load_async_to_lds_b128(
        (__attribute__((address_space(1))) v4i*)gsrc,
        (__attribute__((address_space(3))) v4i*)lds, 0, 0);
#else
    asm volatile("global_load_async_to_lds_b128 %0, %1, off"
                 :: "v"((__attribute__((address_space(3))) v4i*)lds),
                    "v"((__attribute__((address_space(1))) v4i*)gsrc)
                 : "memory");
#endif
}

template <int N>
__device__ __forceinline__ void wait_asynccnt() {
#if __has_builtin(__builtin_amdgcn_s_wait_asynccnt)
    __builtin_amdgcn_s_wait_asynccnt(N);
#else
    asm volatile("s_wait_asynccnt %0" :: "i"(N) : "memory");
#endif
}

// ---------------------------------------------------------------------------
// Kernel 0: x (f32, [B][C][N]) -> Xh (f16, [B][C][N])
// ---------------------------------------------------------------------------
__global__ __launch_bounds__(256)
void cvt_f16(const float* __restrict__ x, _Float16* __restrict__ xh) {
    int i = (blockIdx.x * 256 + threadIdx.x) * 4;
    float4 v = *(const float4*)(x + i);
    H4 h;
    h.h[0] = (_Float16)v.x; h.h[1] = (_Float16)v.y;
    h.h[2] = (_Float16)v.z; h.h[3] = (_Float16)v.w;
    *(uint2*)(xh + i) = __builtin_bit_cast(uint2, h);
}

// ---------------------------------------------------------------------------
// Kernel 1: Q = W1*x + b1, K = W2*x + b2 ; stored TRANSPOSED as
//           Qt,Kt : f16 [B][N][C]
// ---------------------------------------------------------------------------
__global__ __launch_bounds__(256)
void proj_qk(const float* __restrict__ x,
             const float* __restrict__ W1, const float* __restrict__ b1,
             const float* __restrict__ W2, const float* __restrict__ b2,
             _Float16* __restrict__ Qt, _Float16* __restrict__ Kt) {
    __shared__ _Float16 XT[128][136];   // transposed x tile [n_local][ci], padded

    const int b    = blockIdx.y;
    const int n0   = blockIdx.x * 128;
    const int t    = threadIdx.x;
    const int lane = t & 31;
    const int w    = t >> 5;
    const int hi   = lane >> 4;
    const int ml   = lane & 15;

    {
        const int ci = t >> 1;
        const int nh = (t & 1) * 64;
        const float* src = x + ((size_t)b * C_DIM + ci) * N_DIM + n0 + nh;
        #pragma unroll
        for (int j = 0; j < 64; j += 4) {
            float4 v = *(const float4*)(src + j);
            XT[nh + j + 0][ci] = (_Float16)v.x;
            XT[nh + j + 1][ci] = (_Float16)v.y;
            XT[nh + j + 2][ci] = (_Float16)v.z;
            XT[nh + j + 3][ci] = (_Float16)v.w;
        }
    }
    __syncthreads();

    v8f accQ[8], accK[8];
    #pragma unroll
    for (int ct = 0; ct < 8; ++ct) { accQ[ct] = {}; accK[ct] = {}; }

    const int nloc = w * 16 + ml;

    #pragma unroll
    for (int cc = 0; cc < 4; ++cc) {
        const _Float16* bp = &XT[nloc][cc * 32 + hi * 8];
        v16h Bv = make_v16h(*(const uint4*)bp, *(const uint4*)(bp + 16));
        #pragma unroll
        for (int ct = 0; ct < 8; ++ct) {
            const int co = ct * 16 + ml;
            const float* w1r = W1 + (size_t)co * C_DIM + cc * 32 + hi * 8;
            v16h Aq = cvt16(*(const float4*)(w1r),      *(const float4*)(w1r + 4),
                            *(const float4*)(w1r + 16), *(const float4*)(w1r + 20));
            accQ[ct] = wmma_f16(Aq, Bv, accQ[ct]);
            const float* w2r = W2 + (size_t)co * C_DIM + cc * 32 + hi * 8;
            v16h Ak = cvt16(*(const float4*)(w2r),      *(const float4*)(w2r + 4),
                            *(const float4*)(w2r + 16), *(const float4*)(w2r + 20));
            accK[ct] = wmma_f16(Ak, Bv, accK[ct]);
        }
    }

    const size_t rowbase = ((size_t)b * N_DIM + n0 + nloc) * C_DIM;
    #pragma unroll
    for (int ct = 0; ct < 8; ++ct) {
        const float* bb1 = b1 + ct * 16 + hi * 8;
        const float* bb2 = b2 + ct * 16 + hi * 8;
        H8 hq, hk;
        #pragma unroll
        for (int r = 0; r < 8; ++r) {
            hq.h[r] = (_Float16)(accQ[ct][r] + bb1[r]);
            hk.h[r] = (_Float16)(accK[ct][r] + bb2[r]);
        }
        size_t base = rowbase + ct * 16 + hi * 8;
        *(uint4*)(Qt + base) = __builtin_bit_cast(uint4, hq);
        *(uint4*)(Kt + base) = __builtin_bit_cast(uint4, hk);
    }
}

// ---------------------------------------------------------------------------
// Kernel 2: flash attention, K/X tiles double-buffered in LDS via CDNA5
// async global->LDS DMA. Grid (N/128, B), 256 threads = 8 waves; wave w owns
// 16 query rows m = m0 + w*16 + {0..15}.
// ---------------------------------------------------------------------------
__global__ __launch_bounds__(256)
void attn(const _Float16* __restrict__ Qt, const _Float16* __restrict__ Kt,
          const _Float16* __restrict__ Xh, const float* __restrict__ x,
          float* __restrict__ out) {
    __shared__ _Float16 KS[2][NT][136];    // K tile  [n_local][c], padded
    __shared__ _Float16 XS[2][C_DIM][72];  // X tile  [c][n_local], padded
    __shared__ _Float16 PST[8][16][72];    // per-wave P staging [m_local][n_local]
    __shared__ float    RED[8][16];        // per-wave row-stat transpose buffer

    const int b    = blockIdx.y;
    const int m0   = blockIdx.x * 128;
    const int t    = threadIdx.x;
    const int lane = t & 31;
    const int w    = t >> 5;
    const int hi   = lane >> 4;
    const int ml   = lane & 15;
    const int m    = m0 + w * 16 + ml;

    const _Float16* KtB = Kt + (size_t)b * N_DIM * C_DIM;
    const _Float16* XhB = Xh + (size_t)b * C_DIM * N_DIM;

    // issue one n-tile's async copies: 8 x b128 per thread
    auto issue_tile = [&](int nt, int buf) {
        #pragma unroll
        for (int j = 0; j < 4; ++j) {          // K: 64 rows x 16 segs of 16B
            const int segid = t * 4 + j;
            const int row = segid >> 4, seg = segid & 15;
            async_copy16(&KS[buf][row][seg * 8],
                         KtB + (size_t)(nt + row) * C_DIM + seg * 8);
        }
        #pragma unroll
        for (int j = 0; j < 4; ++j) {          // X: 128 rows x 8 segs of 16B
            const int segid = t * 4 + j;
            const int c = segid >> 3, seg = segid & 7;
            async_copy16(&XS[buf][c][seg * 8],
                         XhB + (size_t)c * N_DIM + nt + seg * 8);
        }
    };

    // Q^T A-operand tiles in registers (M=m 16, K=c 32) x 4 chunks
    v16h Qa[4];
    {
        const _Float16* qrow = Qt + ((size_t)b * N_DIM + m) * C_DIM;
        #pragma unroll
        for (int cc = 0; cc < 4; ++cc)
            Qa[cc] = make_v16h(*(const uint4*)(qrow + cc * 32 + hi * 8),
                               *(const uint4*)(qrow + cc * 32 + 16 + hi * 8));
    }

    v8f outAcc[8];
    #pragma unroll
    for (int ct = 0; ct < 8; ++ct) outAcc[ct] = {};
    float rmax[8], rsum[8];
    #pragma unroll
    for (int r = 0; r < 8; ++r) { rmax[r] = -3.0e38f; rsum[r] = 0.0f; }

    issue_tile(0, 0);

    constexpr int ITERS = N_DIM / NT;
    for (int it = 0; it < ITERS; ++it) {
        const int buf = it & 1;
        if (it + 1 < ITERS) {
            issue_tile((it + 1) * NT, buf ^ 1);
            wait_asynccnt<8>();     // tile 'it' done; next tile may be in flight
        } else {
            wait_asynccnt<0>();
        }
        __syncthreads();            // publish tile 'it' to all waves

        // ---- S = Q^T K, 4 n-subtiles of 16 ----
        v8f s[4];
        #pragma unroll
        for (int sub = 0; sub < 4; ++sub) {
            v8f acc = {};
            #pragma unroll
            for (int cc = 0; cc < 4; ++cc) {
                const _Float16* kp = &KS[buf][sub * 16 + ml][cc * 32 + hi * 8];
                v16h Bv = make_v16h(*(const uint4*)kp, *(const uint4*)(kp + 16));
                acc = wmma_f16(Qa[cc], Bv, acc);
            }
            s[sub] = acc;
        }

        // ---- online softmax (rows live within 16-lane halves) ----
        float tmax[8], alpha[8], rs[8];
        #pragma unroll
        for (int r = 0; r < 8; ++r) {
            tmax[r] = fmaxf(fmaxf(s[0][r], s[1][r]), fmaxf(s[2][r], s[3][r]));
            #pragma unroll
            for (int msk = 8; msk; msk >>= 1)
                tmax[r] = fmaxf(tmax[r], __shfl_xor(tmax[r], msk, 32));
            float nm = fmaxf(rmax[r], tmax[r]);
            alpha[r] = __expf(rmax[r] - nm);
            rmax[r]  = nm;
            rs[r]    = 0.0f;
        }
        #pragma unroll
        for (int sub = 0; sub < 4; ++sub)
            #pragma unroll
            for (int r = 0; r < 8; ++r) {
                float p = __expf(s[sub][r] - rmax[r]);
                rs[r] += p;
                PST[w][hi * 8 + r][sub * 16 + ml] = (_Float16)p;
            }
        #pragma unroll
        for (int r = 0; r < 8; ++r) {
            #pragma unroll
            for (int msk = 8; msk; msk >>= 1)
                rs[r] += __shfl_xor(rs[r], msk, 32);
            rsum[r] = rsum[r] * alpha[r] + rs[r];
            if (ml == r) RED[w][hi * 8 + r] = alpha[r];
        }
        const float a = RED[w][ml];
        #pragma unroll
        for (int ct = 0; ct < 8; ++ct)
            #pragma unroll
            for (int r = 0; r < 8; ++r)
                outAcc[ct][r] *= a;

        // ---- O += X * P^T ----
        #pragma unroll
        for (int nch = 0; nch < 2; ++nch) {
            const _Float16* pp = &PST[w][ml][nch * 32 + hi * 8];
            v16h Bv = make_v16h(*(const uint4*)pp, *(const uint4*)(pp + 16));
            #pragma unroll
            for (int ct = 0; ct < 8; ++ct) {
                const _Float16* xr = &XS[buf][ct * 16 + ml][nch * 32];
                v16h Av = make_v16h(*(const uint4*)(xr + hi * 8),
                                    *(const uint4*)(xr + 16 + hi * 8));
                outAcc[ct] = wmma_f16(Av, Bv, outAcc[ct]);
            }
        }

        __syncthreads();            // all waves done with buf before overwrite
    }

    // ---- finalize: /rowsum, +x residual, relu ----
    #pragma unroll
    for (int r = 0; r < 8; ++r)
        if (ml == r) RED[w][hi * 8 + r] = rsum[r];
    const float inv = 1.0f / RED[w][ml];
    #pragma unroll
    for (int ct = 0; ct < 8; ++ct)
        #pragma unroll
        for (int r = 0; r < 8; ++r) {
            const int c = ct * 16 + hi * 8 + r;
            const size_t idx = ((size_t)b * C_DIM + c) * N_DIM + m;
            float v = outAcc[ct][r] * inv + x[idx];
            out[idx] = fmaxf(v, 0.0f);
        }
}

// ---------------------------------------------------------------------------
extern "C" void kernel_launch(void* const* d_in, const int* in_sizes, int n_in,
                              void* d_out, int out_size, void* d_ws, size_t ws_size,
                              hipStream_t stream) {
    const float* x  = (const float*)d_in[0];
    const float* W1 = (const float*)d_in[1];
    const float* b1 = (const float*)d_in[2];
    const float* W2 = (const float*)d_in[3];
    const float* b2 = (const float*)d_in[4];
    float* out = (float*)d_out;

    const size_t elems = (size_t)B_DIM * C_DIM * N_DIM;
    _Float16* Qt = (_Float16*)d_ws;
    _Float16* Kt = Qt + elems;
    _Float16* Xh = Kt + elems;

    cvt_f16<<<dim3((unsigned)(elems / (4 * 256))), dim3(256), 0, stream>>>(x, Xh);
    proj_qk<<<dim3(N_DIM / 128, B_DIM), dim3(256), 0, stream>>>(x, W1, b1, W2, b2, Qt, Kt);
    attn<<<dim3(N_DIM / 128, B_DIM), dim3(256), 0, stream>>>(Qt, Kt, Xh, x, out);
}
